// Bool_39230231281903
// MI455X (gfx1250) — compile-verified, matrix operand order checked by reading, not run on hardware
//
#include <hip/hip_runtime.h>
#include <stdint.h>

// ---------------- types ----------------
typedef __attribute__((ext_vector_type(16))) __bf16 v16bf;
typedef __attribute__((ext_vector_type(8)))  float  v8f;
typedef unsigned int u32x4 __attribute__((ext_vector_type(4)));
typedef int          i32x8 __attribute__((ext_vector_type(8)));
typedef int          i32x4 __attribute__((ext_vector_type(4)));

#define D             128
#define TILE_ROWS     128          // rows per block-tile (8 waves x 16 rows)
#define XROW_STRIDE   132          // floats per LDS x-row (16B pad -> no bank conflicts)
#define WFRAG_STRIDE  10           // dwords per lane inside one B-fragment block (conflict-free b64)
#define WBLK          (32 * WFRAG_STRIDE)          // 320 dwords per (branch,part,kc,nt)
#define W_LDS_DWORDS  (2 * 2 * 4 * 8 * WBLK)       // 40960 dwords = 160 KB
#define W_LDS_BYTES   (W_LDS_DWORDS * 4)
#define XBUF_FLOATS   (TILE_ROWS * XROW_STRIDE)    // 16896 floats = 67584 B
#define LDS_BYTES     (W_LDS_BYTES + 2 * XBUF_FLOATS * 4)   // 299008 B <= 320KB

union AFrag { v16bf v; unsigned u[8]; };

__device__ __forceinline__ v8f wmma_bf16(const AFrag& a, const AFrag& b, v8f c) {
  return __builtin_amdgcn_wmma_f32_16x16x32_bf16(false, a.v, false, b.v, (short)0, c, false, false);
}

// hi = truncated-to-bf16(x); lo = bf16(x - hi): 3-term split recovers ~fp32 GEMM accuracy
__device__ __forceinline__ unsigned pack_hi(float a, float b) {
  return (__float_as_uint(a) >> 16) | (__float_as_uint(b) & 0xFFFF0000u);
}
__device__ __forceinline__ unsigned pack_lo(float a, float b) {
  float ra = a - __uint_as_float(__float_as_uint(a) & 0xFFFF0000u);
  float rb = b - __uint_as_float(__float_as_uint(b) & 0xFFFF0000u);
  return (__float_as_uint(ra) >> 16) | (__float_as_uint(rb) & 0xFFFF0000u);
}

// One TDM descriptor moves a whole 128x128 fp32 tile -> LDS, padding 16B per 512B row.
__device__ __forceinline__ void tdm_load_tile(const float* gsrc, unsigned lds_byte_off) {
  unsigned long long ga = (unsigned long long)(uintptr_t)gsrc;
  u32x4 g0;
  g0[0] = 1u;                                     // count=1, user descriptor
  g0[1] = lds_byte_off;                           // LDS dest (bytes)
  g0[2] = (unsigned)ga;                           // global_addr[31:0]
  g0[3] = (unsigned)((ga >> 32) & 0x01FFFFFFull)  // global_addr[56:32]
        | (2u << 30);                             // type=2 ("image")
  i32x8 g1;
  g1[0] = (int)((2u << 16)      // data_size = 4B
        |       (1u << 20)      // pad_enable
        |       (6u << 22)      // pad_interval: 128 dwords (one 512B row)
        |       (3u << 25));    // pad_amount: 4 dwords (16B)
  g1[1] = (int)(128u << 16);    // tensor_dim0 = 128 elements / row
  g1[2] = (int)(128u << 16);    // tensor_dim1 = 128 rows
  g1[3] = (int)(128u << 16);    // tile_dim0  = 128
  g1[4] = 128;                  // tile_dim1  = 128 rows, tile_dim2 = 0
  g1[5] = 128;                  // tensor_dim0_stride = 128 elements
  g1[6] = 0;
  g1[7] = 0;
  i32x4 z4 = {0, 0, 0, 0};
  i32x8 z8 = {0, 0, 0, 0, 0, 0, 0, 0};
  __builtin_amdgcn_tensor_load_to_lds(g0, g1, z4, z4, z8, 0);
}

__global__ __launch_bounds__(256, 1)
void moe_dual_branch_kernel(const float* __restrict__ x,
                            const float* __restrict__ W0,
                            const float* __restrict__ b0,
                            const float* __restrict__ W1,
                            const float* __restrict__ b1,
                            float* __restrict__ out,
                            int nTiles) {
  extern __shared__ unsigned char smem[];
  unsigned* wl  = (unsigned*)smem;                          // packed B fragments (hi/lo, 2 branches)
  float*    xb0 = (float*)(smem + W_LDS_BYTES);
  float*    xb1 = xb0 + XBUF_FLOATS;

  const int tid  = threadIdx.x;
  const int lane = tid & 31;
  const int wv   = tid >> 5;
  const int m    = lane & 15;       // row within 16-row strip (both lane halves)
  const int kh   = lane >> 4;       // K-half selector for A / row-half for C
  const int col  = m;               // output column within 16-wide tile

  // ---- pack W0/W1 into LDS as bf16 hi/lo B-operand fragments (once per block) ----
  // layout: [branch][part][kc][nt] block of (lane*10 + j) dwords; dword j holds W[k],W[k+1] @ column n
  for (int idx = tid; idx < 32768; idx += 256) {
    int j   =  idx        & 7;
    int ln  = (idx >> 3)  & 31;
    int nt  = (idx >> 8)  & 7;
    int kc  = (idx >> 11) & 3;
    int p   = (idx >> 13) & 1;
    int br  = (idx >> 14) & 1;
    int n   = nt * 16 + (ln & 15);
    int k   = 32 * kc + 2 * (j & 3) + 16 * (j >> 2) + 8 * (ln >> 4);
    const float* W = br ? W1 : W0;
    float f0 = W[k * D + n];
    float f1 = W[(k + 1) * D + n];
    unsigned val = p ? pack_lo(f0, f1) : pack_hi(f0, f1);
    wl[(((br * 2 + p) * 4 + kc) * 8 + nt) * WBLK + ln * WFRAG_STRIDE + j] = val;
  }

  // biases held in registers: col fixed per lane
  float b0r[8], b1r[8];
#pragma unroll
  for (int nt = 0; nt < 8; ++nt) {
    b0r[nt] = b0[nt * 16 + col];
    b1r[nt] = b1[nt * 16 + col];
  }

  const unsigned lds_base = (unsigned)(uintptr_t)smem;   // wave-relative LDS offset (addr[31:0])
  const unsigned xoff0 = lds_base + W_LDS_BYTES;
  const unsigned xoff1 = xoff0 + XBUF_FLOATS * 4;
  const bool wave0 = (wv == 0);
  const int stride = gridDim.x;

  // prologue: TDM prefetch of first tile
  int t0 = blockIdx.x;
  if (wave0 && t0 < nTiles) tdm_load_tile(x + (size_t)t0 * TILE_ROWS * D, xoff0);

  int bufi = 0;
  for (int t = t0; t < nTiles; t += stride, bufi ^= 1) {
    // issue next tile into the other buffer, then wait for current (TDM is in-order per wave)
    if (wave0) {
      int tn = t + stride;
      if (tn < nTiles) {
        tdm_load_tile(x + (size_t)tn * TILE_ROWS * D, bufi ? xoff0 : xoff1);
        __builtin_amdgcn_s_wait_tensorcnt(1);
      } else {
        __builtin_amdgcn_s_wait_tensorcnt(0);
      }
    }
    __syncthreads();   // tile visible to all waves (also covers weight-pack on iter 0)

    const float* xc = bufi ? xb1 : xb0;
    const float* xw = xc + (wv * 16 + m) * XROW_STRIDE;   // this lane's row

    // exact fp32 row-sum for routing (lane l -> row l&15; halves redundant but consistent)
    float s = 0.f;
#pragma unroll
    for (int c = 0; c < D; c += 4) {
      float4 f = *(const float4*)(xw + c);
      s += (f.x + f.y) + (f.z + f.w);
    }
    unsigned rmask = __builtin_amdgcn_ballot_w32(s > 0.0f);

    float* orow = out + (size_t)(t * TILE_ROWS + wv * 16) * D;

    // two passes over 4 n-tiles each: live accumulators = 64 VGPRs/pass,
    // keeps total VGPRs < 256 (no vgpr-msb toggling, 2+ waves/SIMD occupancy)
#pragma unroll
    for (int half = 0; half < 2; ++half) {
      v8f acc0[4], acc1[4];
#pragma unroll
      for (int q = 0; q < 4; ++q) { acc0[q] = (v8f)0.f; acc1[q] = (v8f)0.f; }

#pragma unroll
      for (int kc = 0; kc < 4; ++kc) {
        // build A fragments (hi/lo) straight from padded LDS rows
        AFrag ahi, alo;
#pragma unroll
        for (int j = 0; j < 8; ++j) {
          int k = 32 * kc + 2 * (j & 3) + 16 * (j >> 2) + 8 * kh;
          float2 f = *(const float2*)(xw + k);
          ahi.u[j] = pack_hi(f.x, f.y);
          alo.u[j] = pack_lo(f.x, f.y);
        }
#pragma unroll
        for (int nq = 0; nq < 4; ++nq) {
          const int nt = half * 4 + nq;
          AFrag bh0, bl0, bh1, bl1;
          const unsigned* p00 = wl + ((0 * 4 + kc) * 8 + nt) * WBLK + lane * WFRAG_STRIDE; // br0 hi
          const unsigned* p01 = wl + ((1 * 4 + kc) * 8 + nt) * WBLK + lane * WFRAG_STRIDE; // br0 lo
          const unsigned* p10 = wl + ((2 * 4 + kc) * 8 + nt) * WBLK + lane * WFRAG_STRIDE; // br1 hi
          const unsigned* p11 = wl + ((3 * 4 + kc) * 8 + nt) * WBLK + lane * WFRAG_STRIDE; // br1 lo
#pragma unroll
          for (int q = 0; q < 4; ++q) {
            uint2 a = *(const uint2*)(p00 + 2 * q); bh0.u[2 * q] = a.x; bh0.u[2 * q + 1] = a.y;
            uint2 b = *(const uint2*)(p01 + 2 * q); bl0.u[2 * q] = b.x; bl0.u[2 * q + 1] = b.y;
            uint2 c = *(const uint2*)(p10 + 2 * q); bh1.u[2 * q] = c.x; bh1.u[2 * q + 1] = c.y;
            uint2 d = *(const uint2*)(p11 + 2 * q); bl1.u[2 * q] = d.x; bl1.u[2 * q + 1] = d.y;
          }
          // 3-term split-bf16 GEMM per branch
          acc0[nq] = wmma_bf16(ahi, bh0, acc0[nq]);
          acc0[nq] = wmma_bf16(alo, bh0, acc0[nq]);
          acc0[nq] = wmma_bf16(ahi, bl0, acc0[nq]);
          acc1[nq] = wmma_bf16(ahi, bh1, acc1[nq]);
          acc1[nq] = wmma_bf16(alo, bh1, acc1[nq]);
          acc1[nq] = wmma_bf16(ahi, bl1, acc1[nq]);
        }
      }

      // epilogue for this half: select branch per row, add bias, ReLU, store fp32
#pragma unroll
      for (int nq = 0; nq < 4; ++nq) {
        const int nt = half * 4 + nq;
#pragma unroll
        for (int v = 0; v < 8; ++v) {
          int row = v + 8 * kh;
          int sel = (rmask >> row) & 1;
          float a0 = acc0[nq][v] + b0r[nt];
          float a1 = acc1[nq][v] + b1r[nt];
          float val = sel ? a1 : a0;
          orow[(size_t)row * D + nt * 16 + col] = fmaxf(val, 0.0f);
        }
      }
    }
    __syncthreads();   // all reads of this buffer done before TDM may overwrite it
  }
}

extern "C" void kernel_launch(void* const* d_in, const int* in_sizes, int n_in,
                              void* d_out, int out_size, void* d_ws, size_t ws_size,
                              hipStream_t stream) {
  const float* x  = (const float*)d_in[0];
  const float* W0 = (const float*)d_in[1];
  const float* b0 = (const float*)d_in[2];
  const float* W1 = (const float*)d_in[3];
  const float* b1 = (const float*)d_in[4];
  float* out = (float*)d_out;

  int nTokens = in_sizes[0] / D;
  int nTiles  = nTokens / TILE_ROWS;      // 524288/128 = 4096
  int blocks  = nTiles < 1024 ? nTiles : 1024;

  moe_dual_branch_kernel<<<dim3(blocks), dim3(256), LDS_BYTES, stream>>>(
      x, W0, b0, W1, b1, out, nTiles);
}
